// EquivariantInteractionBlock_15917148799186
// MI455X (gfx1250) — compile-verified, matrix operand order checked by reading, not run on hardware
//
#include <hip/hip_runtime.h>
#include <hip/hip_bf16.h>
#include <stdint.h>

// ---------------------------------------------------------------------------
// CDNA5 (gfx1250) fused equivariant interaction block.
// GEMMs via v_wmma_f32_16x16x32_bf16; weight fragments staged per-workgroup
// into LDS with GLOBAL_LOAD_ASYNC_TO_LDS_B128 (ASYNCcnt-tracked DMA path).
// ---------------------------------------------------------------------------

typedef __bf16 bf16_t;
typedef bf16_t v16bf __attribute__((ext_vector_type(16)));
typedef float  v8f   __attribute__((ext_vector_type(8)));

union FragBits { uint4 q[2]; v16bf v; };

__device__ __forceinline__ unsigned int f32_to_bf16_bits(float x) {
  unsigned int u = __builtin_bit_cast(unsigned int, x);
  return (u + 0x7FFFu + ((u >> 16) & 1u)) >> 16;   // RNE
}
__device__ __forceinline__ unsigned int pack_bf16x2(float lo, float hi) {
  return f32_to_bf16_bits(lo) | (f32_to_bf16_bits(hi) << 16);
}
__device__ __forceinline__ float sigmoidf_(float x) { return 1.0f / (1.0f + __expf(-x)); }

// B-fragment table: frag index = F0 + kt*8 + nt.
// Per-frag layout (512B half-planes to keep ds_load_b128 nearly conflict-free):
//   q0 at frag*256 + lane*4 dwords, q1 at frag*256 + 128 + lane*4 dwords.
// Edge-kernel set [0,120) is contiguous so one linear async copy stages it.
#define F0_W1    0     // K=192 -> 6 k-tiles
#define F0_W2    48    // K=128 -> 4 k-tiles
#define F0_WIN   80
#define F0_WTP   112   // K=16  -> 1 k-tile, zero padded to 32
#define F0_WUP   120
#define F0_WGATE 152
#define NFRAGS   184
#define EDGE_FRAGS 120
#define NODE_FRAGS 64

// A (16x32 bf16) per-lane layout, ISA 7.12.2: lane%16 = M, lane/16 = half.
// VGPR v<4 : K = half*8 + 2v(+1) ; VGPR v>=4 : K = 16 + half*8 + 2(v-4)(+1)
// Rows stored as packed bf16 pairs -> fragment = two contiguous uint4 LDS loads.
__device__ __forceinline__ v16bf load_frag_arow(const unsigned int* rowBase, int kt, int half) {
  FragBits fb;
  fb.q[0] = *(const uint4*)(rowBase + kt * 16 + half * 4);
  fb.q[1] = *(const uint4*)(rowBase + kt * 16 + 8 + half * 4);
  return fb.v;
}
__device__ __forceinline__ v16bf load_frag(const unsigned int* base, int fragIdx, int lane) {
  const unsigned int* p = base + (size_t)fragIdx * 256 + lane * 4;
  FragBits fb;
  fb.q[0] = *(const uint4*)p;
  fb.q[1] = *(const uint4*)(p + 128);
  return fb.v;
}

// Generic pointers to LDS carry the LDS offset in the low 32 bits (aperture check
// truncates the address to those 32 bits).
__device__ __forceinline__ unsigned int lds_addr_of(const void* p) {
  return (unsigned int)(uintptr_t)p;
}

// Async DMA copy global -> LDS, 16B granules, tracked with ASYNCcnt.
__device__ __forceinline__ void async_copy_b128(unsigned int ldsAddr, const void* gaddr) {
  asm volatile("global_load_async_to_lds_b128 %0, %1, off"
               :: "v"(ldsAddr), "v"(gaddr)
               : "memory");
}
__device__ __forceinline__ void wait_asynccnt0() {
  asm volatile("s_wait_asynccnt 0x0" ::: "memory");
}

// ---------------------------------------------------------------------------
__global__ void zero_kernel(float* __restrict__ p, size_t n) {
  size_t i = (size_t)blockIdx.x * blockDim.x + threadIdx.x;
  size_t stride = (size_t)gridDim.x * blockDim.x;
  for (; i < n; i += stride) p[i] = 0.0f;
}

// Pack all weights (row-major [K,128] f32) into bf16 WMMA B-fragments once.
// B layout mirrors A: lane%16 = N column, lane/16 = K half, VGPR pair -> K.
__global__ void pack_weights(const float* __restrict__ W1, const float* __restrict__ W2,
                             const float* __restrict__ W_in, const float* __restrict__ W_gate,
                             const float* __restrict__ W_up, const float* __restrict__ W_tp,
                             unsigned int* __restrict__ frags) {
  int f = blockIdx.x;
  int lane = threadIdx.x;
  const float* W; int K; int f0;
  if (f < F0_W2)        { W = W1;     K = 192; f0 = F0_W1;    }
  else if (f < F0_WIN)  { W = W2;     K = 128; f0 = F0_W2;    }
  else if (f < F0_WTP)  { W = W_in;   K = 128; f0 = F0_WIN;   }
  else if (f < F0_WUP)  { W = W_tp;   K = 16;  f0 = F0_WTP;   }
  else if (f < F0_WGATE){ W = W_up;   K = 128; f0 = F0_WUP;   }
  else                  { W = W_gate; K = 128; f0 = F0_WGATE; }
  int fl = f - f0;
  int kt = fl >> 3, nt = fl & 7;
  int n = nt * 16 + (lane & 15);
  int half = lane >> 4;
  unsigned int d[8];
#pragma unroll
  for (int v = 0; v < 8; ++v) {
    int klocal = (v < 4) ? (half * 8 + 2 * v) : (16 + half * 8 + 2 * (v - 4));
    int k0 = kt * 32 + klocal;
    float x0 = (k0 < K)     ? W[(size_t)k0 * 128 + n]       : 0.0f;
    float x1 = (k0 + 1 < K) ? W[(size_t)(k0 + 1) * 128 + n] : 0.0f;
    d[v] = pack_bf16x2(x0, x1);
  }
  unsigned int* dst = frags + (size_t)f * 256 + lane * 4;
  *(uint4*)dst         = make_uint4(d[0], d[1], d[2], d[3]);
  *(uint4*)(dst + 128) = make_uint4(d[4], d[5], d[6], d[7]);
}

// ---------------------------------------------------------------------------
// Edge kernel: 4 waves/block, one 16-edge tile per wave. 120 WMMAs per tile.
__global__ __launch_bounds__(128) void edge_kernel(
    const float* __restrict__ h, const float* __restrict__ edge_feat,
    const float* __restrict__ sh, const int* __restrict__ edge_i,
    const int* __restrict__ edge_j, const unsigned int* __restrict__ frags,
    const float* __restrict__ b_in, const float* __restrict__ b1,
    const float* __restrict__ b2, float* __restrict__ aggS,
    float* __restrict__ aggE, long E_) {
  __shared__ __align__(16) unsigned int wfrag[EDGE_FRAGS * 256]; // 120 KB bf16 weight frags
  __shared__ __align__(16) unsigned int stage[4][16][100];       // 16x192 bf16 rows, stride 200
  __shared__ __align__(16) unsigned int shst[4][16][16];         // 16x32 bf16 (sh, K-padded)
  const int tid  = threadIdx.x;
  const int lane = tid & 31;
  const int w    = tid >> 5;

  // ---- per-workgroup async DMA of the edge weight set into LDS (ASYNCcnt path)
  {
    unsigned int ldsBase = lds_addr_of(&wfrag[0]);
    const char* g = (const char*)frags;
    for (int i = tid; i < EDGE_FRAGS * 64; i += 128) {   // 64 x b128 per frag
      async_copy_b128(ldsBase + (unsigned int)i * 16u, g + (size_t)i * 16);
    }
    wait_asynccnt0();
  }
  __syncthreads();

  const long tileBase = ((long)blockIdx.x * 4 + w) * 16;
  if (tileBase >= E_) return;                 // wave-uniform: EXEC stays all-ones
  const int half = lane >> 4;
  const int nl = lane & 15;

  // ---- stage x = [h_j | edge_feat] rows as packed bf16 (uniform trip count)
  for (int idx = lane; idx < 16 * 96; idx += 32) {
    int r = idx / 96, c = idx % 96;
    long e = tileBase + r; if (e >= E_) e = E_ - 1;
    int col = 2 * c;
    float x0, x1;
    if (col < 128) {
      long nj = (long)edge_j[e];
      x0 = h[nj * 128 + col]; x1 = h[nj * 128 + col + 1];
    } else {
      x0 = edge_feat[e * 64 + (col - 128)]; x1 = edge_feat[e * 64 + (col - 127)];
    }
    stage[w][r][c] = pack_bf16x2(x0, x1);
  }
  for (int idx = lane; idx < 16 * 16; idx += 32) {
    int r = idx / 16, c = idx % 16;
    long e = tileBase + r; if (e >= E_) e = E_ - 1;
    shst[w][r][c] = (c < 8) ? pack_bf16x2(sh[e * 16 + 2 * c], sh[e * 16 + 2 * c + 1]) : 0u;
  }

  const unsigned int* arow = &stage[w][nl][0];
  const unsigned int* srow = &shst[w][nl][0];

  // ---- equivariant path: eq_msg = (h_j @ W_in + b_in) * (sh @ W_tp)
  v8f eqAcc[8], shAcc[8];
#pragma unroll
  for (int nt = 0; nt < 8; ++nt) { eqAcc[nt] = {}; shAcc[nt] = {}; }
#pragma unroll
  for (int kt = 0; kt < 4; ++kt) {
    v16bf a = load_frag_arow(arow, kt, half);
#pragma unroll
    for (int nt = 0; nt < 8; ++nt) {
      v16bf b = load_frag(wfrag, F0_WIN + kt * 8 + nt, lane);
      eqAcc[nt] = __builtin_amdgcn_wmma_f32_16x16x32_bf16(false, a, false, b, (short)0,
                                                          eqAcc[nt], false, false);
    }
  }
  {
    v16bf a = load_frag_arow(srow, 0, half);
#pragma unroll
    for (int nt = 0; nt < 8; ++nt) {
      v16bf b = load_frag(wfrag, F0_WTP + nt, lane);
      shAcc[nt] = __builtin_amdgcn_wmma_f32_16x16x32_bf16(false, a, false, b, (short)0,
                                                          shAcc[nt], false, false);
    }
  }
#pragma unroll
  for (int nt = 0; nt < 8; ++nt) {
    float bi = b_in[nt * 16 + nl];
    v8f msg = (eqAcc[nt] + bi) * shAcc[nt];
#pragma unroll
    for (int r = 0; r < 8; ++r) {              // D layout: VGPR r -> M = r + 8*half
      long e = tileBase + half * 8 + r;
      if (e < E_) {
        long ni = (long)edge_i[e];
        atomicAdd(&aggE[ni * 128 + nt * 16 + nl], msg[r]);
      }
    }
  }

  // ---- scalar path: t = silu(x @ W1 + b1)
  v8f acc1[8];
#pragma unroll
  for (int nt = 0; nt < 8; ++nt) acc1[nt] = {};
#pragma unroll
  for (int kt = 0; kt < 6; ++kt) {
    v16bf a = load_frag_arow(arow, kt, half);
#pragma unroll
    for (int nt = 0; nt < 8; ++nt) {
      v16bf b = load_frag(wfrag, F0_W1 + kt * 8 + nt, lane);
      acc1[nt] = __builtin_amdgcn_wmma_f32_16x16x32_bf16(false, a, false, b, (short)0,
                                                         acc1[nt], false, false);
    }
  }
  // silu, then write t back into the stage buffer (lane transpose via LDS; DS in-order)
  unsigned short* tbuf = (unsigned short*)&stage[w][0][0];
#pragma unroll
  for (int nt = 0; nt < 8; ++nt) {
    float bb = b1[nt * 16 + nl];
#pragma unroll
    for (int r = 0; r < 8; ++r) {
      float u = acc1[nt][r] + bb;
      float t = u * sigmoidf_(u);
      tbuf[(half * 8 + r) * 200 + nt * 16 + nl] = (unsigned short)f32_to_bf16_bits(t);
    }
  }
  // scalar_msg = t @ W2 + b2 -> atomic segment-sum
  v8f acc2[8];
#pragma unroll
  for (int nt = 0; nt < 8; ++nt) acc2[nt] = {};
#pragma unroll
  for (int kt = 0; kt < 4; ++kt) {
    v16bf a = load_frag_arow(arow, kt, half);
#pragma unroll
    for (int nt = 0; nt < 8; ++nt) {
      v16bf b = load_frag(wfrag, F0_W2 + kt * 8 + nt, lane);
      acc2[nt] = __builtin_amdgcn_wmma_f32_16x16x32_bf16(false, a, false, b, (short)0,
                                                         acc2[nt], false, false);
    }
  }
#pragma unroll
  for (int nt = 0; nt < 8; ++nt) {
    float bb = b2[nt * 16 + nl];
#pragma unroll
    for (int r = 0; r < 8; ++r) {
      long e = tileBase + half * 8 + r;
      if (e < E_) {
        long ni = (long)edge_i[e];
        atomicAdd(&aggS[ni * 128 + nt * 16 + nl], acc2[nt][r] + bb);
      }
    }
  }
}

// ---------------------------------------------------------------------------
// Node kernel: h_new = h + aggS@W_up + b_up ; gate = sigmoid(h_new@W_gate + b_gate)
//              h_eq_new = h_eq + aggE * gate.  64 WMMAs per 16-node tile.
__global__ __launch_bounds__(128) void node_kernel(
    const float* __restrict__ h, const float* __restrict__ h_eq,
    const float* __restrict__ aggS, const float* __restrict__ aggE,
    const unsigned int* __restrict__ frags,
    const float* __restrict__ b_up, const float* __restrict__ b_gate,
    float* __restrict__ out_h, float* __restrict__ out_heq, long Nn) {
  __shared__ __align__(16) unsigned int wfrag[NODE_FRAGS * 256]; // 64 KB: W_up + W_gate
  __shared__ __align__(16) unsigned int stage[4][16][100];
  const int tid  = threadIdx.x;
  const int lane = tid & 31;
  const int w    = tid >> 5;

  {
    unsigned int ldsBase = lds_addr_of(&wfrag[0]);
    const char* g = (const char*)(frags + (size_t)F0_WUP * 256);
    for (int i = tid; i < NODE_FRAGS * 64; i += 128) {
      async_copy_b128(ldsBase + (unsigned int)i * 16u, g + (size_t)i * 16);
    }
    wait_asynccnt0();
  }
  __syncthreads();

  const long tileBase = ((long)blockIdx.x * 4 + w) * 16;
  if (tileBase >= Nn) return;
  const int half = lane >> 4;
  const int nl = lane & 15;

  for (int idx = lane; idx < 16 * 64; idx += 32) {
    int r = idx / 64, c = idx % 64;
    long nd = tileBase + r; if (nd >= Nn) nd = Nn - 1;
    stage[w][r][c] = pack_bf16x2(aggS[nd * 128 + 2 * c], aggS[nd * 128 + 2 * c + 1]);
  }
  const unsigned int* arow = &stage[w][nl][0];

  v8f accU[8];
#pragma unroll
  for (int nt = 0; nt < 8; ++nt) accU[nt] = {};
#pragma unroll
  for (int kt = 0; kt < 4; ++kt) {
    v16bf a = load_frag_arow(arow, kt, half);
#pragma unroll
    for (int nt = 0; nt < 8; ++nt) {
      v16bf b = load_frag(wfrag, kt * 8 + nt, lane);           // W_up local [0,32)
      accU[nt] = __builtin_amdgcn_wmma_f32_16x16x32_bf16(false, a, false, b, (short)0,
                                                         accU[nt], false, false);
    }
  }
  // h_new: residual + bias; store f32 output and bf16 copy into LDS for gate GEMM
  unsigned short* tbuf = (unsigned short*)&stage[w][0][0];
#pragma unroll
  for (int nt = 0; nt < 8; ++nt) {
    float bu = b_up[nt * 16 + nl];
#pragma unroll
    for (int r = 0; r < 8; ++r) {
      long nd = tileBase + half * 8 + r;
      long ndc = (nd < Nn) ? nd : (Nn - 1);
      float v = accU[nt][r] + bu + h[ndc * 128 + nt * 16 + nl];
      if (nd < Nn) out_h[nd * 128 + nt * 16 + nl] = v;
      tbuf[(half * 8 + r) * 200 + nt * 16 + nl] = (unsigned short)f32_to_bf16_bits(v);
    }
  }
  v8f accG[8];
#pragma unroll
  for (int nt = 0; nt < 8; ++nt) accG[nt] = {};
#pragma unroll
  for (int kt = 0; kt < 4; ++kt) {
    v16bf a = load_frag_arow(arow, kt, half);
#pragma unroll
    for (int nt = 0; nt < 8; ++nt) {
      v16bf b = load_frag(wfrag, 32 + kt * 8 + nt, lane);      // W_gate local [32,64)
      accG[nt] = __builtin_amdgcn_wmma_f32_16x16x32_bf16(false, a, false, b, (short)0,
                                                         accG[nt], false, false);
    }
  }
#pragma unroll
  for (int nt = 0; nt < 8; ++nt) {
    float bg = b_gate[nt * 16 + nl];
#pragma unroll
    for (int r = 0; r < 8; ++r) {
      long nd = tileBase + half * 8 + r;
      if (nd < Nn) {
        float g = sigmoidf_(accG[nt][r] + bg);
        long off = nd * 128 + nt * 16 + nl;
        out_heq[off] = h_eq[off] + aggE[off] * g;
      }
    }
  }
}

// ---------------------------------------------------------------------------
extern "C" void kernel_launch(void* const* d_in, const int* in_sizes, int n_in,
                              void* d_out, int out_size, void* d_ws, size_t ws_size,
                              hipStream_t stream) {
  const float* h      = (const float*)d_in[0];
  const float* h_eq   = (const float*)d_in[1];
  const float* ef     = (const float*)d_in[2];
  const float* sh     = (const float*)d_in[3];
  const int*   ei     = (const int*)d_in[4];   // harness: integer -> const int*
  const int*   ej     = (const int*)d_in[5];
  const float* W_in   = (const float*)d_in[6];
  const float* b_in   = (const float*)d_in[7];
  const float* W_gate = (const float*)d_in[8];
  const float* b_gate = (const float*)d_in[9];
  const float* W1     = (const float*)d_in[10];
  const float* b1     = (const float*)d_in[11];
  const float* W2     = (const float*)d_in[12];
  const float* b2     = (const float*)d_in[13];
  const float* W_up   = (const float*)d_in[14];
  const float* b_up   = (const float*)d_in[15];
  const float* W_tp   = (const float*)d_in[16];

  const long Nn = in_sizes[0] / 128;   // N nodes (H=128)
  const long E  = in_sizes[4];         // E edges

  // workspace: [aggS f32 N*128][aggE f32 N*128][bf16 weight fragments]
  float* aggS = (float*)d_ws;
  float* aggE = aggS + (size_t)Nn * 128;
  unsigned int* frags = (unsigned int*)(aggE + (size_t)Nn * 128);

  zero_kernel<<<2048, 256, 0, stream>>>(aggS, (size_t)Nn * 256);
  pack_weights<<<NFRAGS, 32, 0, stream>>>(W1, W2, W_in, W_gate, W_up, W_tp, frags);

  long etiles = (E + 15) / 16;
  long eblocks = (etiles + 3) / 4;
  edge_kernel<<<(int)eblocks, 128, 0, stream>>>(h, ef, sh, ei, ej, frags,
                                                b_in, b1, b2, aggS, aggE, E);
  long ntiles = (Nn + 15) / 16;
  long nblocks = (ntiles + 3) / 4;
  node_kernel<<<(int)nblocks, 128, 0, stream>>>(h, h_eq, aggS, aggE, frags, b_up, b_gate,
                                                (float*)d_out,
                                                (float*)d_out + (size_t)Nn * 128, Nn);
}
